// DeepAttention_8186207666523
// MI455X (gfx1250) — compile-verified, matrix operand order checked by reading, not run on hardware
//
#include <hip/hip_runtime.h>
#include <hip/hip_bf16.h>

typedef _Float16 f16;
typedef __attribute__((ext_vector_type(16))) _Float16 f16x16;
typedef __attribute__((ext_vector_type(8)))  float    f32x8;
typedef __attribute__((ext_vector_type(4)))  float    f32x4;

#define WMMA_F16(a, b, c) \
  __builtin_amdgcn_wmma_f32_16x16x32_f16(false, (a), false, (b), (short)0, (c), false, false)

// ---------------------------------------------------------------------------
// Fragment loader. CDNA5 16-bit A (16x32) layout: lane&15 selects the M row
// (for A) / N column-row of B^T (for B); per lane, halves[0..7] = K0+hi*8..+7,
// halves[8..15] = K0+16+hi*8..+7 where hi = lane>>4. Works for any K-contiguous
// [row][K] array (global or LDS) with 16B-aligned chunks.
// ---------------------------------------------------------------------------
__device__ __forceinline__ f16x16 load_frag(const f16* base, int row, int ld,
                                            int k0, int lane) {
  const f16* p = base + (size_t)row * ld + k0 + ((lane >> 4) << 3);
  f16x16 f;
  *((f32x4*)&f)       = *((const f32x4*)p);
  *(((f32x4*)&f) + 1) = *((const f32x4*)(p + 16));
  return f;
}

__device__ __forceinline__ float sigf(float x) { return 1.0f / (1.0f + __expf(-x)); }

// ---------------------------------------------------------------------------
// Async B-tile staging (CDNA5 TDM-adjacent path): each of the 8 waves issues
// one GLOBAL_LOAD_ASYNC_TO_LDS_B128 (32 lanes x 16B = 512B), together covering
// a 64-row x 32-half (4KB) tile of B^T. Tracked by ASYNCcnt.
// ---------------------------------------------------------------------------
__device__ __forceinline__ void stage_B_async(const f16* Bt, int browbase, int ldb,
                                              int k0, f16* buf, int lane, int wv) {
  int row = wv * 8 + (lane >> 2);
  int ch  = (lane & 3) << 3;
  const f16* g = Bt + (size_t)(browbase + row) * ldb + k0 + ch;
  unsigned lds = (unsigned)(size_t)(buf + row * 32 + ch);
  asm volatile("global_load_async_to_lds_b128 %0, %1, off" : : "v"(lds), "v"(g) : "memory");
}

__device__ __forceinline__ void wait_async_and_barrier() {
  asm volatile("s_wait_asynccnt 0x0" ::: "memory");
  __syncthreads();
}

// Shared wave-GEMM core: 16Mx64N per wave, B double-buffered through LDS.
// ldsB must be a 2*64*32 f16 (8KB) shared buffer; browbase block-uniform.
__device__ __forceinline__ void gemm_tiles(const f16* A, int arow, int lda,
                                           const f16* Bt, int browbase, int ldb,
                                           int K, f16* ldsB, int lane, int wv,
                                           f32x8& c0, f32x8& c1, f32x8& c2, f32x8& c3) {
  stage_B_async(Bt, browbase, ldb, 0, ldsB, lane, wv);
  wait_async_and_barrier();
  int cur = 0;
  for (int k0 = 0; k0 < K; k0 += 32) {
    if (k0 + 32 < K)
      stage_B_async(Bt, browbase, ldb, k0 + 32, ldsB + ((cur ^ 1) << 11), lane, wv);
    if (k0 + 64 < K)
      __builtin_prefetch(A + (size_t)arow * lda + k0 + 64, 0, 1);
    const f16* bb = ldsB + (cur << 11);
    f16x16 a  = load_frag(A, arow, lda, k0, lane);
    f16x16 b0 = load_frag(bb, (lane & 15),      32, 0, lane);
    f16x16 b1 = load_frag(bb, (lane & 15) + 16, 32, 0, lane);
    f16x16 b2 = load_frag(bb, (lane & 15) + 32, 32, 0, lane);
    f16x16 b3 = load_frag(bb, (lane & 15) + 48, 32, 0, lane);
    c0 = WMMA_F16(a, b0, c0);
    c1 = WMMA_F16(a, b1, c1);
    c2 = WMMA_F16(a, b2, c2);
    c3 = WMMA_F16(a, b3, c3);
    wait_async_and_barrier();
    cur ^= 1;
  }
}

// ------------------------------ pack kernels -------------------------------
// x_att = concat(word[300], abstr0[256], abstr1[256]) padded to K=832, f16.
__global__ void pack_att_kernel(const float* __restrict__ word,
                                const float* __restrict__ a0,
                                const float* __restrict__ a1,
                                f16* __restrict__ out) {
  long idx = (long)blockIdx.x * blockDim.x + threadIdx.x;
  if (idx >= 8192L * 832) return;
  int r = (int)(idx / 832), k = (int)(idx % 832);
  float v = 0.0f;
  if (k < 300)      v = word[(size_t)r * 300 + k];
  else if (k < 556) v = a0[(size_t)r * 256 + (k - 300)];
  else if (k < 812) v = a1[(size_t)r * 256 + (k - 556)];
  out[idx] = (f16)v;
}

// W_attn [3,250,812] -> Wh [3,256,832] f16 (zero padded rows/cols)
__global__ void pack_W_kernel(const float* __restrict__ W, f16* __restrict__ Wh) {
  long idx = (long)blockIdx.x * blockDim.x + threadIdx.x;
  if (idx >= 3L * 256 * 832) return;
  int k = (int)(idx % 832);
  int n = (int)((idx / 832) % 256);
  int i = (int)(idx / (832 * 256));
  float v = 0.0f;
  if (n < 250 && k < 812) v = W[((size_t)i * 250 + n) * 812 + k];
  Wh[idx] = (f16)v;
}

// x2_list [3][B=16][512][256] -> x2T [3][16][256][512] f16 (N-major for GEMM B)
__global__ void pack_x2T_kernel(const float* __restrict__ a0,
                                const float* __restrict__ a1,
                                const float* __restrict__ a2,
                                f16* __restrict__ x2T) {
  long idx = (long)blockIdx.x * blockDim.x + threadIdx.x;
  if (idx >= 3L * 16 * 256 * 512) return;
  int m = (int)(idx % 512);
  int d = (int)((idx / 512) % 256);
  int b = (int)((idx / (512 * 256)) % 16);
  int i = (int)(idx / (512L * 256 * 16));
  const float* src = (i == 0) ? a0 : (i == 1) ? a1 : a2;
  x2T[idx] = (f16)src[((size_t)b * 512 + m) * 256 + d];
}

// x1cat[row][0:512] = concat(x1_abstr_0, x1_abstr_1) as f16 (cols 512:1280 come
// from the attention-context GEMM).
__global__ void pack_x1cat_kernel(const float* __restrict__ a0,
                                  const float* __restrict__ a1,
                                  f16* __restrict__ x1cat) {
  long idx = (long)blockIdx.x * blockDim.x + threadIdx.x;
  if (idx >= 8192L * 512) return;
  int r = (int)(idx / 512), c = (int)(idx % 512);
  float v = (c < 256) ? a0[(size_t)r * 256 + c] : a1[(size_t)r * 256 + (c - 256)];
  x1cat[(size_t)r * 1280 + c] = (f16)v;
}

__global__ void cvt_f16_kernel(const float* __restrict__ in, f16* __restrict__ out, long n) {
  long idx = (long)blockIdx.x * blockDim.x + threadIdx.x;
  if (idx < n) out[idx] = (f16)in[idx];
}

// ----------------------- projection GEMM: r = relu(X W^T) (*v) -------------
// grid (64, 4, 6): 8 waves/block * 64 = 512 M-tiles (M=8192), y -> 64-wide N
// group (N=256), z = side*3 + module. B tile shared by all waves via LDS.
__global__ void gemm_r_kernel(const f16* __restrict__ x1att, const f16* __restrict__ x2att,
                              const f16* __restrict__ Wh, const float* __restrict__ v_attn,
                              f16* __restrict__ r1, f16* __restrict__ r2) {
  __shared__ __align__(16) f16 ldsB[2 * 64 * 32];
  int lane = threadIdx.x & 31, wv = threadIdx.x >> 5;
  int m0 = (blockIdx.x * 8 + wv) << 4;
  int n0 = blockIdx.y << 6;
  int side = blockIdx.z / 3, mod = blockIdx.z % 3;
  const f16* A  = side ? x2att : x1att;
  const f16* Bt = Wh + (size_t)mod * 256 * 832;
  f16* out = (side ? r2 : r1) + (size_t)mod * 8192 * 256;
  int arow = m0 + (lane & 15);
  f32x8 c0 = {}, c1 = {}, c2 = {}, c3 = {};
  gemm_tiles(A, arow, 832, Bt, n0, 832, 832, ldsB, lane, wv, c0, c1, c2, c3);
  int nn = lane & 15, mb = (lane >> 4) << 3;
  f32x8 accs[4] = {c0, c1, c2, c3};
#pragma unroll
  for (int t = 0; t < 4; ++t) {
    int n = n0 + t * 16 + nn;
    float scale = 1.0f;
    if (side) scale = (n < 250) ? v_attn[mod * 250 + n] : 0.0f;
#pragma unroll
    for (int j = 0; j < 8; ++j)
      out[(size_t)(m0 + mb + j) * 256 + n] = (f16)(fmaxf(accs[t][j], 0.0f) * scale);
  }
}

// ---------------- batched scores GEMM: sc[b,l,m] = r1[b,l,:].r2[b,m,:] ------
// grid (64, 8): M=8192 (batch b = blockIdx.x>>2, block-uniform), N=512/batch,
// K=256 (cols 250..255 are exact zeros in both operands).
__global__ void score_gemm_kernel(const f16* __restrict__ r1i, const f16* __restrict__ r2i,
                                  float* __restrict__ sc) {
  __shared__ __align__(16) f16 ldsB[2 * 64 * 32];
  int lane = threadIdx.x & 31, wv = threadIdx.x >> 5;
  int m0 = (blockIdx.x * 8 + wv) << 4;
  int b = blockIdx.x >> 2;  // 4 blocks (32 m-tiles) per batch
  int n0 = blockIdx.y << 6;
  int arow = m0 + (lane & 15);
  int browbase = b * 512 + n0;
  f32x8 c0 = {}, c1 = {}, c2 = {}, c3 = {};
  gemm_tiles(r1i, arow, 256, r2i, browbase, 256, 256, ldsB, lane, wv, c0, c1, c2, c3);
  int nn = lane & 15, mb = (lane >> 4) << 3;
  f32x8 accs[4] = {c0, c1, c2, c3};
#pragma unroll
  for (int t = 0; t < 4; ++t)
#pragma unroll
    for (int j = 0; j < 8; ++j)
      sc[(size_t)(m0 + mb + j) * 512 + n0 + t * 16 + nn] = accs[t][j];
}

// -------- masked softmax over L2 (reference uses finfo.min, not -inf) -------
__global__ void softmax_kernel(const float* __restrict__ sc,
                               const unsigned char* __restrict__ x2_mask,
                               f16* __restrict__ alpha) {
  const float NEGF = -3.4028234663852886e38f;
  __shared__ float red[256];
  int row = blockIdx.x, t = threadIdx.x;
  int b = row >> 9;
  float s0 = x2_mask[b * 512 + t]       ? NEGF : sc[(size_t)row * 512 + t];
  float s1 = x2_mask[b * 512 + 256 + t] ? NEGF : sc[(size_t)row * 512 + 256 + t];
  red[t] = fmaxf(s0, s1);
  __syncthreads();
  for (int s = 128; s > 0; s >>= 1) {
    if (t < s) red[t] = fmaxf(red[t], red[t + s]);
    __syncthreads();
  }
  float mx = red[0];
  __syncthreads();
  float e0 = __expf(s0 - mx), e1 = __expf(s1 - mx);
  red[t] = e0 + e1;
  __syncthreads();
  for (int s = 128; s > 0; s >>= 1) {
    if (t < s) red[t] += red[t + s];
    __syncthreads();
  }
  float inv = 1.0f / red[0];
  alpha[(size_t)row * 512 + t]       = (f16)(e0 * inv);
  alpha[(size_t)row * 512 + 256 + t] = (f16)(e1 * inv);
}

// ------- context GEMM: x1cat[:, 512+i*256 : ...] = alpha @ x2_list[i] -------
// grid (64, 4): K=512, N=256, batched via x2T[b] (b block-uniform).
__global__ void pieces_gemm_kernel(const f16* __restrict__ alpha,
                                   const f16* __restrict__ x2Ti,
                                   f16* __restrict__ x1cat, int mod) {
  __shared__ __align__(16) f16 ldsB[2 * 64 * 32];
  int lane = threadIdx.x & 31, wv = threadIdx.x >> 5;
  int m0 = (blockIdx.x * 8 + wv) << 4;
  int b = blockIdx.x >> 2;
  int n0 = blockIdx.y << 6;
  const f16* Bt = x2Ti + (size_t)b * 256 * 512;
  int arow = m0 + (lane & 15);
  f32x8 c0 = {}, c1 = {}, c2 = {}, c3 = {};
  gemm_tiles(alpha, arow, 512, Bt, n0, 512, 512, ldsB, lane, wv, c0, c1, c2, c3);
  int nn = lane & 15, mb = (lane >> 4) << 3;
  f32x8 accs[4] = {c0, c1, c2, c3};
#pragma unroll
  for (int t = 0; t < 4; ++t) {
    int col = 512 + mod * 256 + n0 + t * 16 + nn;
#pragma unroll
    for (int j = 0; j < 8; ++j)
      x1cat[(size_t)(m0 + mb + j) * 1280 + col] = (f16)accs[t][j];
  }
}

// ------------- g_in GEMM: g[dir] = x1cat @ Wih[dir]^T + b[dir] --------------
// grid (64, 8, 2): K=1280, N=512.
__global__ void gin_gemm_kernel(const f16* __restrict__ x1cat, const f16* __restrict__ WihH,
                                const float* __restrict__ b_f, const float* __restrict__ b_b,
                                float* __restrict__ g) {
  __shared__ __align__(16) f16 ldsB[2 * 64 * 32];
  int lane = threadIdx.x & 31, wv = threadIdx.x >> 5;
  int m0 = (blockIdx.x * 8 + wv) << 4;
  int n0 = blockIdx.y << 6;
  int dir = blockIdx.z;
  const f16* Bt = WihH + (size_t)dir * 512 * 1280;
  const float* bias = dir ? b_b : b_f;
  float* out = g + (size_t)dir * 8192 * 512;
  int arow = m0 + (lane & 15);
  f32x8 c0 = {}, c1 = {}, c2 = {}, c3 = {};
  gemm_tiles(x1cat, arow, 1280, Bt, n0, 1280, 1280, ldsB, lane, wv, c0, c1, c2, c3);
  int nn = lane & 15, mb = (lane >> 4) << 3;
  f32x8 accs[4] = {c0, c1, c2, c3};
#pragma unroll
  for (int t = 0; t < 4; ++t) {
    int n = n0 + t * 16 + nn;
#pragma unroll
    for (int j = 0; j < 8; ++j)
      out[(size_t)(m0 + mb + j) * 512 + n] = accs[t][j] + bias[n];
  }
}

// ------------------------------ BiLSTM ------------------------------------
// One workgroup per direction (8 waves). h kept as f16 in LDS and fed to WMMA
// as the A operand (M=16 batches, K=128). Each wave computes 64 of the 512
// gate pre-activations; elementwise phase updates c/h and writes the output.
// B tiles are per-wave here (n0 = wv*64), so direct global loads are kept.
__global__ void lstm_kernel(const f16* __restrict__ WhhH, const float* __restrict__ g,
                            float* __restrict__ out) {
  __shared__ __align__(16) f16 hH[16 * 128];
  __shared__ float cC[16 * 128];
  __shared__ float zb[16 * 512];
  int dir = blockIdx.x;
  const f16* Bt = WhhH + (size_t)dir * 512 * 128;
  const float* gd = g + (size_t)dir * 8192 * 512;
  int t = threadIdx.x, lane = t & 31, wv = t >> 5;
  for (int i = t; i < 2048; i += 256) { hH[i] = (f16)0.0f; cC[i] = 0.0f; }
  __syncthreads();
  int n0 = wv * 64;
  for (int step = 0; step < 512; ++step) {
    int l = dir ? (511 - step) : step;
    f32x8 c0 = {}, c1 = {}, c2 = {}, c3 = {};
    for (int k0 = 0; k0 < 128; k0 += 32) {
      f16x16 a  = load_frag(hH, lane & 15, 128, k0, lane);
      f16x16 b0 = load_frag(Bt, n0 + (lane & 15), 128, k0, lane);
      f16x16 b1 = load_frag(Bt, n0 + 16 + (lane & 15), 128, k0, lane);
      f16x16 b2 = load_frag(Bt, n0 + 32 + (lane & 15), 128, k0, lane);
      f16x16 b3 = load_frag(Bt, n0 + 48 + (lane & 15), 128, k0, lane);
      c0 = WMMA_F16(a, b0, c0);
      c1 = WMMA_F16(a, b1, c1);
      c2 = WMMA_F16(a, b2, c2);
      c3 = WMMA_F16(a, b3, c3);
    }
    int nn = lane & 15, mb = (lane >> 4) << 3;
    f32x8 accs[4] = {c0, c1, c2, c3};
#pragma unroll
    for (int tt = 0; tt < 4; ++tt) {
      int gate = n0 + tt * 16 + nn;
#pragma unroll
      for (int j = 0; j < 8; ++j) {
        int bi = mb + j;
        zb[bi * 512 + gate] = accs[tt][j] + gd[((size_t)bi * 512 + l) * 512 + gate];
      }
    }
    __syncthreads();
    for (int idx = t; idx < 2048; idx += 256) {
      int bi = idx >> 7, h = idx & 127;
      float zi = zb[bi * 512 + h];
      float zf = zb[bi * 512 + 128 + h];
      float zg = zb[bi * 512 + 256 + h];
      float zo = zb[bi * 512 + 384 + h];
      float cc = sigf(zf) * cC[idx] + sigf(zi) * tanhf(zg);
      float hh = sigf(zo) * tanhf(cc);
      cC[idx] = cc;
      hH[idx] = (f16)hh;
      out[((size_t)bi * 512 + l) * 256 + dir * 128 + h] = hh;
    }
    __syncthreads();
  }
}

// ---------------------------------------------------------------------------
extern "C" void kernel_launch(void* const* d_in, const int* in_sizes, int n_in,
                              void* d_out, int out_size, void* d_ws, size_t ws_size,
                              hipStream_t stream) {
  const float* x1_word = (const float*)d_in[0];
  const float* x1_a0   = (const float*)d_in[1];
  const float* x1_a1   = (const float*)d_in[2];
  const float* x2_word = (const float*)d_in[3];
  const float* x2_a0   = (const float*)d_in[4];
  const float* x2_a1   = (const float*)d_in[5];
  const float* x2_a2   = (const float*)d_in[6];
  const unsigned char* x2_mask = (const unsigned char*)d_in[8];
  const float* W_attn  = (const float*)d_in[9];
  const float* v_attn  = (const float*)d_in[10];
  const float* Wih_f   = (const float*)d_in[11];
  const float* Whh_f   = (const float*)d_in[12];
  const float* b_f     = (const float*)d_in[13];
  const float* Wih_b   = (const float*)d_in[14];
  const float* Whh_b   = (const float*)d_in[15];
  const float* b_b     = (const float*)d_in[16];
  float* out = (float*)d_out;

  // Workspace carve-out (~150 MB, deterministic; everything stays L2-resident).
  char* w = (char*)d_ws;
  auto take = [&](size_t bytes) {
    char* p = w;
    w += (bytes + 255) & ~(size_t)255;
    return p;
  };
  f16*   x1att = (f16*)take(8192ull * 832 * 2);
  f16*   x2att = (f16*)take(8192ull * 832 * 2);
  f16*   Wh    = (f16*)take(3ull * 256 * 832 * 2);
  f16*   r1    = (f16*)take(3ull * 8192 * 256 * 2);
  f16*   r2    = (f16*)take(3ull * 8192 * 256 * 2);
  f16*   x2T   = (f16*)take(3ull * 16 * 256 * 512 * 2);
  f16*   x1cat = (f16*)take(8192ull * 1280 * 2);
  f16*   WihH  = (f16*)take(2ull * 512 * 1280 * 2);
  f16*   WhhH  = (f16*)take(2ull * 512 * 128 * 2);
  float* sc    = (float*)take(8192ull * 512 * 4);
  f16*   alpha = (f16*)take(8192ull * 512 * 2);
  float* gg    = (float*)take(2ull * 8192 * 512 * 4);

  // ---- pack / convert ----
  {
    long n = 8192L * 832;
    pack_att_kernel<<<(n + 255) / 256, 256, 0, stream>>>(x1_word, x1_a0, x1_a1, x1att);
    pack_att_kernel<<<(n + 255) / 256, 256, 0, stream>>>(x2_word, x2_a0, x2_a1, x2att);
  }
  pack_W_kernel<<<(3L * 256 * 832 + 255) / 256, 256, 0, stream>>>(W_attn, Wh);
  pack_x2T_kernel<<<(3L * 16 * 256 * 512 + 255) / 256, 256, 0, stream>>>(x2_a0, x2_a1, x2_a2, x2T);
  pack_x1cat_kernel<<<(8192L * 512 + 255) / 256, 256, 0, stream>>>(x1_a0, x1_a1, x1cat);
  cvt_f16_kernel<<<(512L * 1280 + 255) / 256, 256, 0, stream>>>(Wih_f, WihH, 512L * 1280);
  cvt_f16_kernel<<<(512L * 1280 + 255) / 256, 256, 0, stream>>>(Wih_b, WihH + 512L * 1280, 512L * 1280);
  cvt_f16_kernel<<<(512L * 128 + 255) / 256, 256, 0, stream>>>(Whh_f, WhhH, 512L * 128);
  cvt_f16_kernel<<<(512L * 128 + 255) / 256, 256, 0, stream>>>(Whh_b, WhhH + 512L * 128, 512L * 128);

  // ---- projections (both sides, all 3 modules) ----
  gemm_r_kernel<<<dim3(64, 4, 6), 256, 0, stream>>>(x1att, x2att, Wh, v_attn, r1, r2);

  // ---- per-module: scores -> softmax -> context (reusing sc/alpha buffers) ----
  for (int i = 0; i < 3; ++i) {
    score_gemm_kernel<<<dim3(64, 8), 256, 0, stream>>>(r1 + (size_t)i * 8192 * 256,
                                                       r2 + (size_t)i * 8192 * 256, sc);
    softmax_kernel<<<8192, 256, 0, stream>>>(sc, x2_mask, alpha);
    pieces_gemm_kernel<<<dim3(64, 4), 256, 0, stream>>>(alpha, x2T + (size_t)i * 16 * 256 * 512,
                                                        x1cat, i);
  }

  // ---- LSTM input gates + recurrence ----
  gin_gemm_kernel<<<dim3(64, 8, 2), 256, 0, stream>>>(x1cat, WihH, b_f, b_b, gg);
  lstm_kernel<<<2, 256, 0, stream>>>(WhhH, gg, out);
}
